// MultiScaleRetriever_75453985456496
// MI455X (gfx1250) — matrix operand order repaired and checked by reading, not compiled, not run on hardware
//
#include <hip/hip_runtime.h>

// ---------------- problem constants (match reference) ----------------
constexpr int kB      = 1024;     // batch
constexpr int kD      = 1024;     // dim
constexpr int kN      = 100000;   // bank entries
constexpr int kSplit  = 32;       // N-dimension splits for the scores kernel
constexpr int kChunk  = 3136;     // 49 * 64 ; kSplit*kChunk = 100352
constexpr int kNPad   = kSplit * kChunk;  // 100352, multiple of 64
constexpr int kTopM   = 32;
constexpr int kRows   = 32;       // q-rows per scores workgroup (2 WMMA A-tiles)
constexpr int kCand   = kSplit * kTopM;   // candidates per row at merge (1024)
constexpr float kInvTemp = 1.0f / 0.07f;

typedef __attribute__((ext_vector_type(16))) __bf16 v16bf;
typedef __attribute__((ext_vector_type(8)))  float  v8f;

union V16 { uint4 q[2]; v16bf v; };

// ---------------- K0: keys = bf16(l2norm(bank)), zero-padded ----------------
__global__ __launch_bounds__(128) void keys_kernel(const float* __restrict__ bank,
                                                   __bf16* __restrict__ keysb) {
  __shared__ float red[128];
  const int tid = threadIdx.x;
  const int n   = blockIdx.x;
  __bf16* out = keysb + (size_t)n * kD;
  if (n >= kN) {
    for (int d = tid; d < kD; d += 128) out[d] = (__bf16)0.0f;
    return;
  }
  const float* br = bank + (size_t)n * kD;
  float v[8];
  float ss = 0.0f;
#pragma unroll
  for (int r = 0; r < 8; ++r) {
    v[r] = br[tid + 128 * r];
    ss += v[r] * v[r];
  }
  red[tid] = ss;
  __syncthreads();
  for (int st = 64; st > 0; st >>= 1) {
    if (tid < st) red[tid] += red[tid + st];
    __syncthreads();
  }
  const float scale = 1.0f / fmaxf(sqrtf(red[0]), 1e-12f);
#pragma unroll
  for (int r = 0; r < 8; ++r) out[tid + 128 * r] = (__bf16)(v[r] * scale);
}

// ---------------- K1: p = q @ W^T + b ; pn = bf16(l2norm(p)) ----------------
__global__ __launch_bounds__(256) void proj_norm_kernel(const float* __restrict__ q,
                                                        const float* __restrict__ W,
                                                        const float* __restrict__ bias,
                                                        __bf16* __restrict__ pnb) {
  __shared__ float qs[kD];
  __shared__ float red[256];
  const int tid = threadIdx.x;
  const int row = blockIdx.x;
  const float* qr = q + (size_t)row * kD;
  for (int k = tid; k < kD; k += 256) qs[k] = qr[k];
  __syncthreads();
  float p[4];
#pragma unroll
  for (int r = 0; r < 4; ++r) {
    const int j = tid + 256 * r;
    const float4* Wr = (const float4*)(W + (size_t)j * kD);
    float acc = 0.0f;
    for (int k4 = 0; k4 < kD / 4; ++k4) {
      const float4 w  = Wr[k4];
      const float4 qv = ((const float4*)qs)[k4];
      acc += qv.x * w.x + qv.y * w.y + qv.z * w.z + qv.w * w.w;
    }
    p[r] = acc + bias[j];
  }
  float ss = p[0] * p[0] + p[1] * p[1] + p[2] * p[2] + p[3] * p[3];
  red[tid] = ss;
  __syncthreads();
  for (int st = 128; st > 0; st >>= 1) {
    if (tid < st) red[tid] += red[tid + st];
    __syncthreads();
  }
  const float scale = 1.0f / fmaxf(sqrtf(red[0]), 1e-12f);
#pragma unroll
  for (int r = 0; r < 4; ++r)
    pnb[(size_t)row * kD + tid + 256 * r] = (__bf16)(p[r] * scale);
}

// ---------------- K2: scores GEMM (bf16 WMMA) fused with per-row top-32 ----------------
// grid = (kSplit, kB/kRows), block = 128 (4 waves).
// Each wave owns two 16x16 score tiles (rows 0-15 & 16-31 of the WG) sharing one B load.
// Top-32 selection: lanes filter their own accumulator values against a per-row LDS
// threshold, push rare survivors into a per-row queue (LDS atomics); 32 threads drain.
__global__ __launch_bounds__(128) void scores_topk_kernel(const __bf16* __restrict__ pnb,
                                                          const uint4*  __restrict__ keysq,
                                                          float* __restrict__ candS,
                                                          int*   __restrict__ candI) {
  __shared__ uint4 At[2][16][129];       // two 16x1024 bf16 A tiles (padded rows)
  __shared__ float topS[kRows][33];
  __shared__ int   topI[kRows][33];
  __shared__ float thr[kRows];           // current 32nd-best per row
  __shared__ int   cnt[kRows];           // pending-queue counts
  __shared__ float pendS[kRows][64];
  __shared__ int   pendI[kRows][64];

  const int tid  = threadIdx.x;
  const int lane = tid & 31;
  const int wave = tid >> 5;
  const int rowbase   = blockIdx.y * kRows;
  const int chunk     = blockIdx.x;
  const int chunkbase = chunk * kChunk;

  // Stage both A tiles (32 q-rows of pn, bf16) into LDS.
  {
    const uint4* src = (const uint4*)pnb + (size_t)rowbase * (kD / 8);
    for (int e = tid; e < kRows * 128; e += 128) {
      const int rr = e >> 7, cc = e & 127;
      At[rr >> 4][rr & 15][cc] = src[(size_t)rr * (kD / 8) + cc];
    }
  }
  for (int e = tid; e < kRows * kTopM; e += 128) {
    topS[e >> 5][e & 31] = -3.0e38f;
    topI[e >> 5][e & 31] = -1;
  }
  if (tid < kRows) { thr[tid] = -3.0e38f; cnt[tid] = 0; }
  __syncthreads();

  const int arow = lane & 15;
  const int asel = (lane < 16) ? 0 : 1;  // A: K halves {0..7,16..23} / {8..15,24..31}
  const int bsel = (lane < 16) ? 0 : 1;  // B: K halves {0..7,16..23} / {8..15,24..31}
  const int mbase = (lane < 16) ? 0 : 8;

  float rminv = -3.0e38f;  // running min of this row's top-32 (threads 0..31)
  int   rminp = 0;

  for (int it = 0; it < kChunk / 64; ++it) {
    const int nbase = chunkbase + it * 64;
    const int n0    = nbase + wave * 16;
    const uint4* Brow = keysq + (size_t)(n0 + arow) * (kD / 8);
    // speculative prefetch of next tile's key row
    {
      int npf = n0 + 64 + arow;
      if (npf > kNPad - 1) npf = kNPad - 1;
      __builtin_prefetch(keysq + (size_t)npf * (kD / 8), 0, 0);
    }

    v8f acc0 = {};
    v8f acc1 = {};
#pragma unroll 4
    for (int kk = 0; kk < kD; kk += 32) {
      const int u = kk >> 3;  // uint4 index of this K-slice
      V16 a0, a1, b;
      b.q[0]  = Brow[u + bsel];
      b.q[1]  = Brow[u + bsel + 2];
      a0.q[0] = At[0][arow][u + asel];
      a0.q[1] = At[0][arow][u + asel + 2];
      a1.q[0] = At[1][arow][u + asel];
      a1.q[1] = At[1][arow][u + asel + 2];
      acc0 = __builtin_amdgcn_wmma_f32_16x16x32_bf16(false, a0.v, false, b.v,
                                                     (short)0, acc0, false, false);
      acc1 = __builtin_amdgcn_wmma_f32_16x16x32_bf16(false, a1.v, false, b.v,
                                                     (short)0, acc1, false, false);
    }

    // Filter phase: C VGPR r holds M=r (lanes 0-15) / M=r+8 (lanes 16-31), N=lane%16.
    const int colg = nbase + (wave << 4) + (lane & 15);  // global key index
    if (colg < kN) {
#pragma unroll
      for (int r = 0; r < 8; ++r) {
        {
          const int   row = mbase + r;
          const float s   = acc0[r];
          if (s > thr[row]) {
            const int slot = atomicAdd(&cnt[row], 1);
            if (slot < 64) { pendS[row][slot] = s; pendI[row][slot] = colg; }
          }
        }
        {
          const int   row = 16 + mbase + r;
          const float s   = acc1[r];
          if (s > thr[row]) {
            const int slot = atomicAdd(&cnt[row], 1);
            if (slot < 64) { pendS[row][slot] = s; pendI[row][slot] = colg; }
          }
        }
      }
    }
    __syncthreads();

    // Drain phase: one thread per row inserts survivors into its top-32.
    if (tid < kRows) {
      const int row = tid;
      int k = cnt[row];
      if (k > 64) k = 64;
      for (int i = 0; i < k; ++i) {
        const float s  = pendS[row][i];
        const int   gn = pendI[row][i];
        if (s > rminv) {
          topS[row][rminp] = s;
          topI[row][rminp] = gn;
          float mv = topS[row][0];
          int   mp = 0;
#pragma unroll
          for (int j = 1; j < kTopM; ++j) {
            const float t = topS[row][j];
            if (t < mv) { mv = t; mp = j; }
          }
          rminv = mv;
          rminp = mp;
        }
      }
      cnt[row] = 0;
      thr[row] = rminv;
    }
    __syncthreads();
  }

  for (int e = tid; e < kRows * kTopM; e += 128) {
    const int row = e >> 5, m = e & 31;
    const size_t o = (((size_t)(rowbase + row)) * kSplit + chunk) * kTopM + m;
    candS[o] = topS[row][m];
    candI[o] = topI[row][m];
  }
}

// ---------------- K3: merge kCand candidates -> top-32, softmax, weighted gather ----------------
__global__ __launch_bounds__(256) void merge_gather_kernel(const float* __restrict__ candS,
                                                           const int*   __restrict__ candI,
                                                           const float* __restrict__ bank,
                                                           float* __restrict__ z) {
  __shared__ float cs[kCand];
  __shared__ int   ci[kCand];
  __shared__ float selS[32];
  __shared__ int   selI[32];
  __shared__ float attn[32];
  const int tid = threadIdx.x;
  const int row = blockIdx.x;
  const size_t base = (size_t)row * kCand;
  for (int e = tid; e < kCand; e += 256) {
    cs[e] = candS[base + e];
    ci[e] = candI[base + e];
  }
  __syncthreads();
#pragma unroll
  for (int h = 0; h < kCand / 256; ++h) {
    const int c = tid + h * 256;
    const float s  = cs[c];
    const int   ix = ci[c];
    int rank = 0;
    for (int o = 0; o < kCand; ++o) {
      const float so = cs[o];
      rank += (so > s) || (so == s && ci[o] < ix);  // strict total order, no self-count
    }
    if (rank < kTopM) { selS[rank] = s; selI[rank] = ix; }
  }
  __syncthreads();
  if (tid == 0) {
    const float mx = selS[0];
    float sum = 0.0f;
    for (int m = 0; m < kTopM; ++m) {
      const float e = __expf((selS[m] - mx) * kInvTemp);
      attn[m] = e;
      sum += e;
    }
    const float inv = 1.0f / sum;
    for (int m = 0; m < kTopM; ++m) attn[m] *= inv;
  }
  __syncthreads();
  float acc[4] = {0.0f, 0.0f, 0.0f, 0.0f};
  for (int m = 0; m < kTopM; ++m) {
    const float a = attn[m];
    const float* brow = bank + (size_t)selI[m] * kD;
#pragma unroll
    for (int r = 0; r < 4; ++r) acc[r] += a * brow[tid + 256 * r];
  }
#pragma unroll
  for (int r = 0; r < 4; ++r) z[(size_t)row * kD + tid + 256 * r] = acc[r];
}

// ---------------- K4: ref (+)= z @ U^T + ub ----------------
__global__ __launch_bounds__(256) void unify_kernel(const float* __restrict__ z,
                                                    const float* __restrict__ U,
                                                    const float* __restrict__ ub,
                                                    float* __restrict__ refb,
                                                    int accumulate) {
  __shared__ float zs[kD];
  const int tid = threadIdx.x;
  const int row = blockIdx.x;
  const float* zr = z + (size_t)row * kD;
  for (int k = tid; k < kD; k += 256) zs[k] = zr[k];
  __syncthreads();
#pragma unroll
  for (int r = 0; r < 4; ++r) {
    const int j = tid + 256 * r;
    const float4* Ur = (const float4*)(U + (size_t)j * kD);
    float acc = 0.0f;
    for (int k4 = 0; k4 < kD / 4; ++k4) {
      const float4 w  = Ur[k4];
      const float4 qv = ((const float4*)zs)[k4];
      acc += qv.x * w.x + qv.y * w.y + qv.z * w.z + qv.w * w.w;
    }
    acc += ub[j];
    const size_t o = (size_t)row * kD + j;
    refb[o] = accumulate ? (refb[o] + acc) : acc;
  }
}

// ---------------- K5: layernorm ----------------
__global__ __launch_bounds__(256) void layernorm_kernel(const float* __restrict__ refb,
                                                        const float* __restrict__ gamma,
                                                        const float* __restrict__ beta,
                                                        float* __restrict__ out) {
  __shared__ float red[256];
  __shared__ float red2[256];
  const int tid = threadIdx.x;
  const int row = blockIdx.x;
  const float* xr = refb + (size_t)row * kD;
  float s = 0.0f, s2 = 0.0f;
  float v[4];
#pragma unroll
  for (int r = 0; r < 4; ++r) {
    v[r] = xr[tid + 256 * r];
    s  += v[r];
    s2 += v[r] * v[r];
  }
  red[tid]  = s;
  red2[tid] = s2;
  __syncthreads();
  for (int st = 128; st > 0; st >>= 1) {
    if (tid < st) { red[tid] += red[tid + st]; red2[tid] += red2[tid + st]; }
    __syncthreads();
  }
  const float mean = red[0] * (1.0f / kD);
  const float var  = red2[0] * (1.0f / kD) - mean * mean;
  const float inv  = rsqrtf(var + 1e-5f);
#pragma unroll
  for (int r = 0; r < 4; ++r) {
    const int d = tid + 256 * r;
    out[(size_t)row * kD + d] = (v[r] - mean) * inv * gamma[d] + beta[d];
  }
}

// ---------------- launch ----------------
extern "C" void kernel_launch(void* const* d_in, const int* in_sizes, int n_in,
                              void* d_out, int out_size, void* d_ws, size_t ws_size,
                              hipStream_t stream) {
  (void)in_sizes; (void)n_in; (void)out_size; (void)ws_size;
  const float* q       = (const float*)d_in[0];
  const float* bank    = (const float*)d_in[1];
  const float* proj_w  = (const float*)d_in[2];
  const float* proj_b  = (const float*)d_in[3];
  const float* unify_w = (const float*)d_in[4];
  const float* unify_b = (const float*)d_in[5];
  const float* gamma   = (const float*)d_in[6];
  const float* beta    = (const float*)d_in[7];
  float* out = (float*)d_out;

  char* ws = (char*)d_ws;
  size_t off = 0;
  auto alloc = [&](size_t bytes) -> char* {
    off = (off + 255) & ~(size_t)255;
    char* p = ws + off;
    off += bytes;
    return p;
  };
  __bf16* keysb = (__bf16*)alloc((size_t)kNPad * kD * 2);   // ~196 MB
  __bf16* pnb   = (__bf16*)alloc((size_t)kB * kD * 2);      // 2 MB
  float*  candS = (float*)alloc((size_t)kB * kSplit * kTopM * 4);
  int*    candI = (int*)  alloc((size_t)kB * kSplit * kTopM * 4);
  float*  z     = (float*)alloc((size_t)kB * kD * 4);
  float*  refb  = (float*)alloc((size_t)kB * kD * 4);

  keys_kernel<<<kNPad, 128, 0, stream>>>(bank, keysb);

  for (int s = 0; s < 2; ++s) {
    proj_norm_kernel<<<kB, 256, 0, stream>>>(q, proj_w + (size_t)s * kD * kD,
                                             proj_b + (size_t)s * kD, pnb);
    dim3 g2(kSplit, kB / kRows);
    scores_topk_kernel<<<g2, 128, 0, stream>>>(pnb, (const uint4*)keysb, candS, candI);
    merge_gather_kernel<<<kB, 256, 0, stream>>>(candS, candI, bank, z);
    unify_kernel<<<kB, 256, 0, stream>>>(z, unify_w + (size_t)s * kD * kD,
                                         unify_b + (size_t)s * kD, refb, s);
  }
  layernorm_kernel<<<kB, 256, 0, stream>>>(refb, gamma, beta, out);
}
